// Block_50130858279051
// MI455X (gfx1250) — matrix operand order repaired and checked by reading, not compiled, not run on hardware
//
#include <hip/hip_runtime.h>
#include <hip/hip_bf16.h>

// ---------------------------------------------------------------------------
// Transformer block for MI455X (gfx1250, wave32, WMMA + TDM).
//   x + attn(ln1(x)) -> x1 ; x1 + mlp(ln2(x1)) -> out
// GEMMs: fp16 in / f32 accum on v_wmma_f32_16x16x32_f16.
// Weight tiles are DMA'd to LDS by the Tensor Data Mover (tensor_load_to_lds)
// with hardware row padding; activations are staged with b128 copies.
// ---------------------------------------------------------------------------

typedef __attribute__((ext_vector_type(16))) _Float16 v16h;
typedef __attribute__((ext_vector_type(8)))  float    v8f;
typedef __attribute__((ext_vector_type(4)))  unsigned v4u;
typedef __attribute__((ext_vector_type(8)))  int      v8i;
typedef __attribute__((ext_vector_type(4)))  int      v4i;

#define CDIM 1024
#define TDIM 2048
#define BDIM 2
#define NH   16
#define HD   64
#define ROWS (BDIM*TDIM)   // 4096

// CDNA5 fp16 A-matrix (16x32) element->K mapping: lane half (0-15 / 16-31)
// offsets K by 0/8; VGPRs 0..3 hold K 0..7(+off), VGPRs 4..7 hold K 16..23(+off).
__device__ __forceinline__ int kA_of(int i, int lane) {
  return 2*i + ((i >= 4) ? 8 : 0) + ((lane & 16) ? 8 : 0);
}
// CDNA5 fp16 B-matrix (32x16): same K mapping as A: lane half offsets K by 0/8;
// VGPRs 0..3 hold K 0..7(+off), VGPRs 4..7 hold K 16..23(+off).
__device__ __forceinline__ int kB_of(int i, int lane) {
  return 2*i + ((i >= 4) ? 8 : 0) + ((lane & 16) ? 8 : 0);
}

// ---------------------------------------------------------------------------
// TDM: DMA a 2-D tile (tile_dim0=32 halves along K, tile_dim1=128 rows along N)
// from a row-major [N][K] fp16 tensor into LDS, padding each 16-dword row with
// 4 dwords so the LDS row stride is 40 halves (bank-spread fragment reads).
// Descriptor packing per CDNA5 ISA.
__device__ __forceinline__ void tdm_load_b_tile(unsigned lds_addr,
                                                unsigned long long gaddr,
                                                int K, int N) {
  v4u g0;
  g0[0] = 1u;                                     // count=1, user descriptor
  g0[1] = lds_addr;                               // LDS byte address
  g0[2] = (unsigned)gaddr;                        // global_addr[31:0]
  g0[3] = (unsigned)((gaddr >> 32) & 0x01FFFFFFu) // global_addr[56:32]
          | (2u << 30);                           // type=2 ("image")
  v8i g1;
  g1[0] = (1 << 16)                               // data_size = 2 bytes
        | (1 << 20)                               // pad_enable
        | (3 << 22)                               // pad_interval: 16 dwords
        | (3 << 25);                              // pad_amount:   4 dwords
  g1[1] = (int)(((unsigned)K & 0xFFFFu) << 16);   // tensor_dim0[15:0]
  g1[2] = (int)(((unsigned)K >> 16) | (((unsigned)N & 0xFFFFu) << 16));
  g1[3] = (int)(((unsigned)N >> 16) | (32u << 16));   // tile_dim0 = 32
  g1[4] = 128;                                    // tile_dim1 = 128, tile_dim2 = 0
  g1[5] = K;                                      // tensor_dim0_stride[31:0]
  g1[6] = 0;
  g1[7] = 0;
  v4i z4; z4[0]=0; z4[1]=0; z4[2]=0; z4[3]=0;     // 2-D tensor: groups 2/3 unused
#if __clang_major__ >= 23
  v8i z8; for (int i = 0; i < 8; ++i) z8[i] = 0;
  __builtin_amdgcn_tensor_load_to_lds(g0, g1, z4, z4, z8, 0);
#else
  __builtin_amdgcn_tensor_load_to_lds(g0, g1, z4, z4, 0);
#endif
}

// ---------------------------------------------------------------------------
// Fused f32->f16 convert + transpose: in[K][N] f32 -> out[N][K] f16.
// 32x32 tiles through LDS, coalesced on both sides.
__global__ __launch_bounds__(256)
void convert_transpose_kernel(const float* __restrict__ in,
                              _Float16* __restrict__ out, int K, int N) {
  __shared__ float tile[32][33];
  int nb = blockIdx.x * 32, kb = blockIdx.y * 32;
  int tx = threadIdx.x & 31, ty = threadIdx.x >> 5;   // 32 x 8
  #pragma unroll
  for (int r = ty; r < 32; r += 8)
    tile[r][tx] = in[(size_t)(kb + r) * N + nb + tx];
  __syncthreads();
  #pragma unroll
  for (int r = ty; r < 32; r += 8)
    out[(size_t)(nb + r) * K + kb + tx] = (_Float16)tile[tx][r];
}

// ---------------------------------------------------------------------------
// One row (C=1024) per 256-thread block. f32 in -> f16 normalized out.
__global__ __launch_bounds__(256)
void layernorm_f16_kernel(const float* __restrict__ x,
                          const float* __restrict__ g,
                          const float* __restrict__ b,
                          _Float16* __restrict__ out) {
  int row = blockIdx.x;
  int tid = threadIdx.x;
  const float* xr = x + (size_t)row * CDIM;
  float4 v = ((const float4*)xr)[tid];
  float s = v.x + v.y + v.z + v.w;
  float q = v.x*v.x + v.y*v.y + v.z*v.z + v.w*v.w;
  for (int off = 16; off; off >>= 1) {
    s += __shfl_xor(s, off);
    q += __shfl_xor(q, off);
  }
  __shared__ float rs[8], rq[8];
  int lane = tid & 31, wid = tid >> 5;
  if (lane == 0) { rs[wid] = s; rq[wid] = q; }
  __syncthreads();
  float ts = 0.f, tq = 0.f;
  for (int w = 0; w < 8; ++w) { ts += rs[w]; tq += rq[w]; }
  float mu   = ts * (1.0f / CDIM);
  float var  = tq * (1.0f / CDIM) - mu * mu;
  float rstd = rsqrtf(var + 1e-5f);
  float4 gv = ((const float4*)g)[tid];
  float4 bv = ((const float4*)b)[tid];
  _Float16* o = out + (size_t)row * CDIM + tid * 4;
  o[0] = (_Float16)((v.x - mu) * rstd * gv.x + bv.x);
  o[1] = (_Float16)((v.y - mu) * rstd * gv.y + bv.y);
  o[2] = (_Float16)((v.z - mu) * rstd * gv.z + bv.z);
  o[3] = (_Float16)((v.w - mu) * rstd * gv.w + bv.w);
}

// ---------------------------------------------------------------------------
// WMMA GEMM: C[M,N] = A[M,K](f16,row-major) * BT[N,K](f16,row-major)^T + bias.
//   EPI 0: out f16 | EPI 1: exact GELU -> f16 | EPI 2: + resid(f32) -> f32
// Block tile 128x128x32, 256 threads = 8 waves (2 along M x 4 along N);
// each wave owns 64x32 = 8 WMMA f32 accumulators. A staged with b128 copies,
// BT tile DMA'd by the TDM with hardware padding to a 40-half row stride.
template<int EPI>
__global__ __launch_bounds__(256)
void gemm_wmma_kernel(const _Float16* __restrict__ A,
                      const _Float16* __restrict__ BT,
                      const float* __restrict__ bias,
                      const float* __restrict__ resid,
                      _Float16* __restrict__ outh,
                      float* __restrict__ outf,
                      int M, int N, int K) {
  __shared__ __align__(16) _Float16 sA[128 * 40];  // [m][k], rows padded to 40
  __shared__ __align__(16) _Float16 sB[128 * 40];  // [n][k], rows padded to 40 (TDM)

  int tid = threadIdx.x, lane = tid & 31, wid = tid >> 5;
  int wm = wid >> 2, wn = wid & 3;
  int m0 = blockIdx.y * 128, n0 = blockIdx.x * 128;

  v8f acc[4][2];
  for (int mt = 0; mt < 4; ++mt)
    for (int nt = 0; nt < 2; ++nt)
      for (int r = 0; r < 8; ++r) acc[mt][nt][r] = 0.f;

  int arow = tid >> 1, apart = tid & 1;   // A staging: 2 threads per row
  unsigned sB_lds = (unsigned)(unsigned long long)(uintptr_t)sB;
  unsigned long long bt_base = (unsigned long long)(uintptr_t)BT;

  for (int k0 = 0; k0 < K; k0 += 32) {
    // ---- B tile (128 x 32) via Tensor Data Mover ----
    if (tid < 32) {
      tdm_load_b_tile(sB_lds,
                      bt_base + ((size_t)n0 * K + k0) * sizeof(_Float16), K, N);
    }
    // ---- stage A tile (128 x 32) ----
    {
      const _Float16* ag = A + (size_t)(m0 + arow) * K + k0 + apart * 16;
      uint4 a0 = ((const uint4*)ag)[0];
      uint4 a1 = ((const uint4*)ag)[1];
      *((uint4*)(sA + arow * 40 + apart * 16))     = a0;
      *((uint4*)(sA + arow * 40 + apart * 16 + 8)) = a1;
    }
    if (k0 + 32 < K)
      __builtin_prefetch(A + (size_t)(m0 + arow) * K + k0 + 32, 0, 1);
    if (tid < 32) __builtin_amdgcn_s_wait_tensorcnt(0);
    __syncthreads();

    // ---- fragments from LDS ----
    v16h af[4], bf[2];
    #pragma unroll
    for (int mt = 0; mt < 4; ++mt) {
      int m = wm * 64 + mt * 16 + (lane & 15);
      unsigned* u = (unsigned*)&af[mt];
      #pragma unroll
      for (int i = 0; i < 8; ++i)
        u[i] = *(const unsigned*)((const char*)sA + (m * 40 + kA_of(i, lane)) * 2);
    }
    #pragma unroll
    for (int nt = 0; nt < 2; ++nt) {
      int n = wn * 32 + nt * 16 + (lane & 15);
      unsigned* u = (unsigned*)&bf[nt];
      #pragma unroll
      for (int i = 0; i < 8; ++i)
        u[i] = *(const unsigned*)((const char*)sB + (n * 40 + kB_of(i, lane)) * 2);
    }
    #pragma unroll
    for (int mt = 0; mt < 4; ++mt)
      #pragma unroll
      for (int nt = 0; nt < 2; ++nt)
        acc[mt][nt] = __builtin_amdgcn_wmma_f32_16x16x32_f16(
            false, af[mt], false, bf[nt], (short)0, acc[mt][nt], false, false);
    __syncthreads();
  }

  // ---- epilogue ----
  int half8 = (lane & 16) ? 8 : 0;
  #pragma unroll
  for (int mt = 0; mt < 4; ++mt) {
    #pragma unroll
    for (int nt = 0; nt < 2; ++nt) {
      int n = n0 + wn * 32 + nt * 16 + (lane & 15);
      #pragma unroll
      for (int r = 0; r < 8; ++r) {
        int m = m0 + wm * 64 + mt * 16 + r + half8;
        float v = acc[mt][nt][r] + bias[n];
        if (EPI == 1) v = 0.5f * v * (1.0f + erff(v * 0.70710678118f));
        if (EPI == 2) {
          v += resid[(size_t)m * N + n];
          outf[(size_t)m * N + n] = v;
        } else {
          outh[(size_t)m * N + n] = (_Float16)v;
        }
      }
    }
  }
}

// ---------------------------------------------------------------------------
// Causal flash attention. Grid (T/128, H, B), 256 threads = 8 waves.
// Each wave owns 16 query rows; iterates 32-key blocks with online softmax.
// qkv layout: [B, T, 3C] fp16 (q | k | v), head h at columns h*64..h*64+63.
__global__ __launch_bounds__(256)
void attn_kernel(const _Float16* __restrict__ qkv, _Float16* __restrict__ y) {
  __shared__ __align__(16) _Float16 sP[8][16 * 40];  // per-wave P tile
  __shared__ __align__(16) _Float16 sV[8][32 * 72];  // per-wave V block [t][d]

  int lane = threadIdx.x & 31, wid = threadIdx.x >> 5;
  int h = blockIdx.y, b = blockIdx.z;
  int q0 = blockIdx.x * 128 + wid * 16;
  const _Float16* base = qkv + (size_t)b * TDIM * 3 * CDIM;
  const int RS = 3 * CDIM;
  int half8 = (lane & 16) ? 8 : 0;
  int l15   = lane & 15;

  // Q as two A-fragments (hd=64 -> K 0..31 and 32..63)
  v16h qa[2];
  {
    const _Float16* qrow = base + (size_t)(q0 + l15) * RS + h * HD;
    #pragma unroll
    for (int kh = 0; kh < 2; ++kh) {
      unsigned* u = (unsigned*)&qa[kh];
      #pragma unroll
      for (int i = 0; i < 8; ++i)
        u[i] = *(const unsigned*)(qrow + kh * 32 + kA_of(i, lane));
    }
  }

  float mrow[8], lrow[8];
  v8f o[4];
  for (int r = 0; r < 8; ++r) { mrow[r] = -1e30f; lrow[r] = 0.f; }
  for (int nt = 0; nt < 4; ++nt)
    for (int r = 0; r < 8; ++r) o[nt][r] = 0.f;

  int nblk = (q0 + 16 + 31) >> 5;     // causal: keys < q0+16
  for (int jb = 0; jb < nblk; ++jb) {
    int j0 = jb * 32;

    // stage V block (32 keys x 64 dims) into this wave's LDS region
    {
      const _Float16* vrow = base + (size_t)(j0 + lane) * RS + 2 * CDIM + h * HD;
      _Float16* dst = &sV[wid][lane * 72];
      #pragma unroll
      for (int c = 0; c < 8; ++c)
        *((uint4*)(dst + c * 8)) = *((const uint4*)(vrow + c * 8));
    }

    // scores: two 16-key tiles, each = two chained WMMAs over hd
    float sc[2][8];
    #pragma unroll
    for (int kt = 0; kt < 2; ++kt) {
      int key = j0 + kt * 16 + l15;
      const _Float16* krow = base + (size_t)key * RS + CDIM + h * HD;
      v16h kb0, kb1;
      unsigned* u0 = (unsigned*)&kb0;
      unsigned* u1 = (unsigned*)&kb1;
      #pragma unroll
      for (int i = 0; i < 8; ++i) {
        int d = kB_of(i, lane);
        u0[i] = *(const unsigned*)(krow + d);
        u1[i] = *(const unsigned*)(krow + 32 + d);
      }
      v8f s;
      for (int r = 0; r < 8; ++r) s[r] = 0.f;
      s = __builtin_amdgcn_wmma_f32_16x16x32_f16(false, qa[0], false, kb0, (short)0, s, false, false);
      s = __builtin_amdgcn_wmma_f32_16x16x32_f16(false, qa[1], false, kb1, (short)0, s, false, false);
      #pragma unroll
      for (int r = 0; r < 8; ++r) {
        float v = s[r] * 0.125f;                 // 1/sqrt(64)
        int qq  = q0 + r + half8;
        sc[kt][r] = (key > qq) ? -1e30f : v;     // causal mask (data, not control)
      }
    }

    // online softmax per row (row = lane-half x accumulator index)
    #pragma unroll
    for (int r = 0; r < 8; ++r) {
      float v = fmaxf(sc[0][r], sc[1][r]);
      v = fmaxf(v, __shfl_xor(v, 1)); v = fmaxf(v, __shfl_xor(v, 2));
      v = fmaxf(v, __shfl_xor(v, 4)); v = fmaxf(v, __shfl_xor(v, 8));
      float mnew  = fmaxf(mrow[r], v);
      float alpha = __expf(mrow[r] - mnew);
      mrow[r] = mnew;
      float p0 = __expf(sc[0][r] - mnew);
      float p1 = __expf(sc[1][r] - mnew);
      float ps = p0 + p1;
      ps += __shfl_xor(ps, 1); ps += __shfl_xor(ps, 2);
      ps += __shfl_xor(ps, 4); ps += __shfl_xor(ps, 8);
      lrow[r] = lrow[r] * alpha + ps;
      #pragma unroll
      for (int nt = 0; nt < 4; ++nt) o[nt][r] *= alpha;
      int mm = r + half8;
      sP[wid][mm * 40 + l15]      = (_Float16)p0;
      sP[wid][mm * 40 + 16 + l15] = (_Float16)p1;
    }
    asm volatile("s_wait_dscnt 0" ::: "memory");

    // P as an A-fragment (16 queries x 32 keys)
    v16h pa;
    {
      unsigned* u = (unsigned*)&pa;
      #pragma unroll
      for (int i = 0; i < 8; ++i)
        u[i] = *(const unsigned*)((const char*)&sP[wid][0] + (l15 * 40 + kA_of(i, lane)) * 2);
    }
    // V B-fragments (32 keys x 16 dims each), accumulate O
    #pragma unroll
    for (int nt = 0; nt < 4; ++nt) {
      v16h vb;
      _Float16* e = (_Float16*)&vb;
      int n = nt * 16 + l15;
      #pragma unroll
      for (int i = 0; i < 8; ++i) {
        int k = kB_of(i, lane);
        e[2 * i]     = sV[wid][k * 72 + n];
        e[2 * i + 1] = sV[wid][(k + 1) * 72 + n];
      }
      o[nt] = __builtin_amdgcn_wmma_f32_16x16x32_f16(false, pa, false, vb, (short)0, o[nt], false, false);
    }
  }

  // normalize and write y (fp16, feeds the proj GEMM)
  #pragma unroll
  for (int nt = 0; nt < 4; ++nt) {
    #pragma unroll
    for (int r = 0; r < 8; ++r) {
      int mm = r + half8;
      float val = o[nt][r] / lrow[r];
      y[((size_t)b * TDIM + q0 + mm) * CDIM + h * HD + nt * 16 + l15] = (_Float16)val;
    }
  }
}

// ---------------------------------------------------------------------------
extern "C" void kernel_launch(void* const* d_in, const int* in_sizes, int n_in,
                              void* d_out, int out_size, void* d_ws, size_t ws_size,
                              hipStream_t stream) {
  (void)in_sizes; (void)n_in; (void)out_size; (void)ws_size;
  const float* x      = (const float*)d_in[0];
  const float* ln1_g  = (const float*)d_in[1];
  const float* ln1_b  = (const float*)d_in[2];
  const float* attn_w = (const float*)d_in[3];
  const float* attn_b = (const float*)d_in[4];
  const float* proj_w = (const float*)d_in[5];
  const float* proj_b = (const float*)d_in[6];
  const float* ln2_g  = (const float*)d_in[7];
  const float* ln2_b  = (const float*)d_in[8];
  const float* fc_w   = (const float*)d_in[9];
  const float* fc_b   = (const float*)d_in[10];
  const float* fc2_w  = (const float*)d_in[11];
  const float* fc2_b  = (const float*)d_in[12];

  char* p = (char*)d_ws;
  auto carve_h = [&](size_t n) { _Float16* r = (_Float16*)p; p += ((n * 2 + 255) / 256) * 256; return r; };
  auto carve_f = [&](size_t n) { float*    r = (float*)p;    p += ((n * 4 + 255) / 256) * 256; return r; };

  // transposed fp16 weights: [N][K]
  _Float16* attn_wT = carve_h((size_t)3 * CDIM * CDIM);
  _Float16* proj_wT = carve_h((size_t)CDIM * CDIM);
  _Float16* fc_wT   = carve_h((size_t)4 * CDIM * CDIM);
  _Float16* fc2_wT  = carve_h((size_t)4 * CDIM * CDIM);
  _Float16* ln1h    = carve_h((size_t)ROWS * CDIM);
  _Float16* qkvh    = carve_h((size_t)ROWS * 3 * CDIM);
  _Float16* yh      = carve_h((size_t)ROWS * CDIM);
  float*    x1      = carve_f((size_t)ROWS * CDIM);
  _Float16* ln2h    = carve_h((size_t)ROWS * CDIM);
  _Float16* hh      = carve_h((size_t)ROWS * 4 * CDIM);

  // one-time (per launch) weight convert+transpose, coalesced via LDS tiles
  convert_transpose_kernel<<<dim3(3 * CDIM / 32, CDIM / 32), 256, 0, stream>>>(
      attn_w, attn_wT, CDIM, 3 * CDIM);
  convert_transpose_kernel<<<dim3(CDIM / 32, CDIM / 32), 256, 0, stream>>>(
      proj_w, proj_wT, CDIM, CDIM);
  convert_transpose_kernel<<<dim3(4 * CDIM / 32, CDIM / 32), 256, 0, stream>>>(
      fc_w, fc_wT, CDIM, 4 * CDIM);
  convert_transpose_kernel<<<dim3(CDIM / 32, 4 * CDIM / 32), 256, 0, stream>>>(
      fc2_w, fc2_wT, 4 * CDIM, CDIM);

  // ln1 -> qkv -> attention -> proj(+x residual, f32 out)
  layernorm_f16_kernel<<<ROWS, 256, 0, stream>>>(x, ln1_g, ln1_b, ln1h);
  gemm_wmma_kernel<0><<<dim3(3 * CDIM / 128, ROWS / 128), 256, 0, stream>>>(
      ln1h, attn_wT, attn_b, nullptr, qkvh, nullptr, ROWS, 3 * CDIM, CDIM);
  attn_kernel<<<dim3(TDIM / 128, NH, BDIM), 256, 0, stream>>>(qkvh, yh);
  gemm_wmma_kernel<2><<<dim3(CDIM / 128, ROWS / 128), 256, 0, stream>>>(
      yh, proj_wT, proj_b, x, nullptr, x1, ROWS, CDIM, CDIM);

  // ln2 -> fc1(+gelu) -> fc2(+x1 residual, f32 out -> d_out)
  layernorm_f16_kernel<<<ROWS, 256, 0, stream>>>(x1, ln2_g, ln2_b, ln2h);
  gemm_wmma_kernel<1><<<dim3(4 * CDIM / 128, ROWS / 128), 256, 0, stream>>>(
      ln2h, fc_wT, fc_b, nullptr, hh, nullptr, ROWS, 4 * CDIM, CDIM);
  gemm_wmma_kernel<2><<<dim3(CDIM / 128, ROWS / 128), 256, 0, stream>>>(
      hh, fc2_wT, fc2_b, x1, nullptr, (float*)d_out, ROWS, CDIM, 4 * CDIM);
}
